// VoxelFeatureExtractor_15719580304498
// MI455X (gfx1250) — matrix operand order repaired and checked by reading, not compiled, not run on hardware
//
#include <hip/hip_runtime.h>
#include <hip/hip_bf16.h>

typedef __attribute__((ext_vector_type(16))) _Float16 v16h;
typedef __attribute__((ext_vector_type(8)))  _Float16 v8h;
typedef __attribute__((ext_vector_type(8)))  float    v8f;

#define KVOX 30000
#define TPT  35
#define MROWS (KVOX * TPT)   // 1,050,000 rows; multiple of 16 -> no M padding
#define NBSTAT 512

// ---------------- per-voxel mean of xyz ----------------
__global__ void k_mean(const float* __restrict__ feat, const int* __restrict__ nv,
                       float* __restrict__ mean3) {
  int k = blockIdx.x * blockDim.x + threadIdx.x;
  if (k >= KVOX) return;
  const float* f = feat + (size_t)k * TPT * 4;
  float s0 = 0.f, s1 = 0.f, s2 = 0.f;
  for (int t = 0; t < TPT; ++t) { s0 += f[t*4+0]; s1 += f[t*4+1]; s2 += f[t*4+2]; }
  float inv = 1.0f / (float)nv[k];
  mean3[k*3+0] = s0*inv; mean3[k*3+1] = s1*inv; mean3[k*3+2] = s2*inv;
}

// ---------------- layer1 GEMM 7->16 (tiny K: VALU) ----------------
__global__ void k_gemm1(const float* __restrict__ feat, const float* __restrict__ mean3,
                        const float* __restrict__ w1, float* __restrict__ y1) {
  __shared__ float sw[112];
  if (threadIdx.x < 112) sw[threadIdx.x] = w1[threadIdx.x];
  __syncthreads();
  int m = blockIdx.x * blockDim.x + threadIdx.x;
  if (m >= MROWS) return;
  int k = m / TPT;
  const float* f = feat + (size_t)m * 4;
  float x[7];
  x[0]=f[0]; x[1]=f[1]; x[2]=f[2]; x[3]=f[3];
  x[4]=f[0]-mean3[k*3+0]; x[5]=f[1]-mean3[k*3+1]; x[6]=f[2]-mean3[k*3+2];
  float* yp = y1 + (size_t)m * 16;
#pragma unroll
  for (int c = 0; c < 16; ++c) {
    float a = 0.f;
#pragma unroll
    for (int j = 0; j < 7; ++j) a += x[j] * sw[c*7+j];
    yp[c] = a;
  }
}

// ---------------- standalone stats (layer 1 only; y1 is small) ----------------
template<int C>
__global__ void k_stats_partial(const float* __restrict__ y, float* __restrict__ part) {
  const int G = 256 / C;
  int c   = threadIdx.x % C;
  int sub = threadIdx.x / C;
  int chunk = (MROWS + gridDim.x - 1) / gridDim.x;
  int r0 = blockIdx.x * chunk;
  int r1 = min(r0 + chunk, MROWS);
  float s = 0.f, q = 0.f;
  for (int r = r0 + sub; r < r1; r += G) {
    float v = y[(size_t)r * C + c];
    s += v; q += v * v;
  }
  __shared__ float ss[256], sq[256];
  ss[threadIdx.x] = s; sq[threadIdx.x] = q;
  __syncthreads();
  if (sub == 0) {
    for (int g = 1; g < G; ++g) { s += ss[g*C + c]; q += sq[g*C + c]; }
    part[(size_t)blockIdx.x * 2 * C + c]     = s;
    part[(size_t)blockIdx.x * 2 * C + C + c] = q;
  }
}

// ---------------- mid-stage: reduce per-GEMM-block partials to NBSTAT rows ----------------
__global__ void k_stats_mid(const float* __restrict__ part1, long nb1, int C2,
                            float* __restrict__ part2) {
  long chunk = (nb1 + gridDim.x - 1) / gridDim.x;
  long b0 = (long)blockIdx.x * chunk;
  long b1 = b0 + chunk; if (b1 > nb1) b1 = nb1;
  int c = threadIdx.x;
  if (c >= C2) return;
  float s = 0.f;
  for (long b = b0; b < b1; ++b) s += part1[(size_t)b * C2 + c];
  part2[(size_t)blockIdx.x * C2 + c] = s;
}

// ---------------- final: fold stats + gamma/beta into per-channel scale/shift ----------------
__global__ void k_stats_final(const float* __restrict__ part, int nb, int C,
                              const float* __restrict__ gamma, const float* __restrict__ beta,
                              float* __restrict__ scsh) {
  int c = threadIdx.x;
  if (c >= C) return;
  float s = 0.f, q = 0.f;
  for (int b = 0; b < nb; ++b) {
    s += part[(size_t)b * 2 * C + c];
    q += part[(size_t)b * 2 * C + C + c];
  }
  const float invN = 1.0f / (float)MROWS;
  float mean = s * invN;
  float var  = q * invN - mean * mean;   // biased var, matches jnp.var
  float scale = gamma[c] * rsqrtf(var + 1e-3f);
  scsh[2*c]   = scale;
  scsh[2*c+1] = beta[c] - mean * scale;
}

// ---------------- f32 -> f16 weight conversion ----------------
__global__ void k_cvt_f16(const float* __restrict__ src, _Float16* __restrict__ dst, int n) {
  int i = blockIdx.x * blockDim.x + threadIdx.x;
  if (i < n) dst[i] = (_Float16)src[i];
}

// ---------------- BN + ReLU + max-over-T + concat + mask (layer1 -> x32 f16) ----------------
__global__ void k_build32(const float* __restrict__ y1, const float* __restrict__ scsh,
                          const int* __restrict__ nv, _Float16* __restrict__ x32) {
  int tid = threadIdx.x;
  int wid = tid >> 5, lane = tid & 31;
  int laneLo = lane & 15, laneHi = lane >> 4;
  int k = blockIdx.x * 16 + wid * 2 + laneHi;   // 2 voxels per wave, 16 per block
  if (k >= KVOX) return;
  int c = laneLo;
  float scale = scsh[2*c], shift = scsh[2*c+1];
  int n = nv[k];
  float mx = -1e30f;
  size_t base = (size_t)k * TPT;
  for (int t = 0; t < TPT; ++t) {            // max over ALL T (mask applied after)
    float v = fmaxf(y1[(base+t)*16 + c] * scale + shift, 0.0f);
    mx = fmaxf(mx, v);
    x32[(base+t)*32 + c] = (_Float16)((t < n) ? v : 0.0f);
  }
  for (int t = 0; t < TPT; ++t)
    x32[(base+t)*32 + 16 + c] = (_Float16)((t < n) ? mx : 0.0f);
}

// ---------------- same for layer2 (64ch -> x128 f16) ----------------
__global__ void k_build128(const float* __restrict__ y2, const float* __restrict__ scsh,
                           const int* __restrict__ nv, _Float16* __restrict__ x128) {
  int tid = threadIdx.x;
  int wid = tid >> 5, lane = tid & 31;
  int k = blockIdx.x * 8 + wid;                 // 1 voxel per wave
  if (k >= KVOX) return;
  int c0 = lane, c1 = lane + 32;
  float s0 = scsh[2*c0], h0 = scsh[2*c0+1];
  float s1 = scsh[2*c1], h1 = scsh[2*c1+1];
  int n = nv[k];
  float m0 = -1e30f, m1 = -1e30f;
  size_t base = (size_t)k * TPT;
  for (int t = 0; t < TPT; ++t) {
    float v0 = fmaxf(y2[(base+t)*64 + c0] * s0 + h0, 0.0f);
    float v1 = fmaxf(y2[(base+t)*64 + c1] * s1 + h1, 0.0f);
    m0 = fmaxf(m0, v0); m1 = fmaxf(m1, v1);
    bool valid = t < n;
    x128[(base+t)*128 + c0] = (_Float16)(valid ? v0 : 0.0f);
    x128[(base+t)*128 + c1] = (_Float16)(valid ? v1 : 0.0f);
  }
  for (int t = 0; t < TPT; ++t) {
    bool valid = t < n;
    x128[(base+t)*128 + 64 + c0] = (_Float16)(valid ? m0 : 0.0f);
    x128[(base+t)*128 + 64 + c1] = (_Float16)(valid ? m1 : 0.0f);
  }
}

// ---------------- WMMA GEMM with fused BN-stat partials ----------------
// Y[M,ND] = A[M,KD](f16) * W[ND,KD]^T (f32 accum), plus per-block per-channel
// partial sum/sumsq of Y reduced from the f32 accumulators (saves a full HBM
// re-read of Y for the batch-norm statistics).
// A layout per lane (16-bit 16x32 A): row = lane&15, halves = K 16*(lane>=16)+h (contiguous)
// B layout per lane (16-bit 32x16 B): col = lane&15, halves = K 16*(lane>=16)+h (contiguous)
template<int KD, int ND>
__global__ void k_gemm_wmma(const _Float16* __restrict__ A,
                            const _Float16* __restrict__ W,
                            float* __restrict__ Y,
                            float* __restrict__ part) {
  const int NT = ND / 16;
  __shared__ float red[2][8][16];
  int tid = threadIdx.x;
  int wid = tid >> 5, lane = tid & 31;
  int laneLo = lane & 15, laneHi = lane >> 4;
  long gw = (long)blockIdx.x * 8 + wid;       // one 16x16 tile per wave
  const long total = (long)(MROWS / 16) * NT;
  bool active = gw < total;

  float s = 0.f, q = 0.f;
  if (active) {
    int  nt   = (int)(gw % NT);
    long row0 = (gw / NT) * 16;
    const _Float16* ap = A + (size_t)(row0 + laneLo) * KD;
    const _Float16* wp = W + (size_t)(nt * 16 + laneLo) * KD + 16 * laneHi;
    v8f c = {0.f, 0.f, 0.f, 0.f, 0.f, 0.f, 0.f, 0.f};
#pragma unroll
    for (int ks = 0; ks < KD / 32; ++ks) {
      union { v16h v; v8h h[2]; } a, b;
      a.h[0] = *(const v8h*)(ap + 32*ks + 16*laneHi);
      a.h[1] = *(const v8h*)(ap + 32*ks + 16*laneHi + 8);
      b.h[0] = *(const v8h*)(wp + 32*ks);
      b.h[1] = *(const v8h*)(wp + 32*ks + 8);
      c = __builtin_amdgcn_wmma_f32_16x16x32_f16(false, a.v, false, b.v,
                                                 (short)0, c, false, false);
    }
    // C/D layout: element r -> row row0 + r + 8*laneHi, col nt*16 + laneLo
    float* yp = Y + (size_t)(row0 + 8*laneHi) * ND + nt*16 + laneLo;
#pragma unroll
    for (int r = 0; r < 8; ++r) {
      yp[(size_t)r * ND] = c[r];
      s += c[r]; q += c[r] * c[r];
    }
  }
  // lane l and l+16 hold the same output column: combine halves
  s += __shfl_xor(s, 16, 32);
  q += __shfl_xor(q, 16, 32);

  ((float*)red)[tid] = 0.f;   // 2*8*16 == 256 floats, zero inactive-wave slots
  __syncthreads();
  if (active && laneHi == 0) { red[0][wid][laneLo] = s; red[1][wid][laneLo] = q; }
  __syncthreads();
  // block covers channel c via waves with (w % NT) == c/16 (blockIdx*8 % NT == 0)
  if (tid < ND) {
    int c = tid;
    float ps = 0.f, pq = 0.f;
#pragma unroll
    for (int w = 0; w < 8; ++w) {
      if ((w % NT) == (c >> 4)) { ps += red[0][w][c & 15]; pq += red[1][w][c & 15]; }
    }
    part[(size_t)blockIdx.x * (2 * ND) + c]      = ps;
    part[(size_t)blockIdx.x * (2 * ND) + ND + c] = pq;
  }
}

// ---------------- final BN + ReLU + masked max-over-T ----------------
__global__ void k_out(const float* __restrict__ y3, const float* __restrict__ scsh,
                      const int* __restrict__ nv, float* __restrict__ out) {
  int tid = threadIdx.x;
  int wid = tid >> 5, lane = tid & 31;
  int k = blockIdx.x * 8 + wid;
  if (k >= KVOX) return;
  int n = nv[k];
  float sc[4], sh[4], m[4];
#pragma unroll
  for (int j = 0; j < 4; ++j) {
    int c = lane + 32*j; sc[j] = scsh[2*c]; sh[j] = scsh[2*c+1]; m[j] = 0.0f;
  }
  size_t base = (size_t)k * TPT;
  for (int t = 0; t < n; ++t) {      // masked rows contribute 0; ReLU>=0 so init 0 is exact
    const float* yp = y3 + (base + t) * 128;
#pragma unroll
    for (int j = 0; j < 4; ++j) {
      float v = fmaxf(yp[lane + 32*j] * sc[j] + sh[j], 0.0f);
      m[j] = fmaxf(m[j], v);
    }
  }
#pragma unroll
  for (int j = 0; j < 4; ++j) out[(size_t)k * 128 + lane + 32*j] = m[j];
}

extern "C" void kernel_launch(void* const* d_in, const int* in_sizes, int n_in,
                              void* d_out, int out_size, void* d_ws, size_t ws_size,
                              hipStream_t stream) {
  const float* feat = (const float*)d_in[0];
  const int*   nv   = (const int*)d_in[1];
  // d_in[2] coors: unused by reference output
  const float* w1 = (const float*)d_in[3];
  const float* g1 = (const float*)d_in[4];
  const float* b1 = (const float*)d_in[5];
  const float* w2 = (const float*)d_in[6];
  const float* g2 = (const float*)d_in[7];
  const float* b2 = (const float*)d_in[8];
  const float* w3 = (const float*)d_in[9];
  const float* g3 = (const float*)d_in[10];
  const float* b3 = (const float*)d_in[11];
  float* out = (float*)d_out;
  (void)in_sizes; (void)n_in; (void)out_size; (void)ws_size;

  const long tiles2  = (long)(MROWS / 16) * (64 / 16);
  const long tiles3  = (long)(MROWS / 16) * (128 / 16);
  const long blocks2 = (tiles2 + 7) / 8;
  const long blocks3 = (tiles3 + 7) / 8;

  // workspace carve-out (~940 MB): y buffer reused for y1/y2/y3
  char* p = (char*)d_ws;
  auto carve = [&](size_t bytes) -> char* {
    char* r = p; p += (bytes + 255) & ~(size_t)255; return r;
  };
  float*    ybuf  = (float*)   carve((size_t)MROWS * 128 * sizeof(float));
  _Float16* x32   = (_Float16*)carve((size_t)MROWS * 32  * sizeof(_Float16));
  _Float16* x128  = (_Float16*)carve((size_t)MROWS * 128 * sizeof(_Float16));
  float*    part1 = (float*)   carve((size_t)blocks3 * 256 * sizeof(float)); // GEMM partials (max)
  float*    part2 = (float*)   carve((size_t)NBSTAT * 256 * sizeof(float));
  float*    mean3 = (float*)   carve((size_t)KVOX * 3 * sizeof(float));
  float*    scsh1 = (float*)   carve(32  * sizeof(float));
  float*    scsh2 = (float*)   carve(128 * sizeof(float));
  float*    scsh3 = (float*)   carve(256 * sizeof(float));
  _Float16* w2h   = (_Float16*)carve(64 * 32 * sizeof(_Float16));
  _Float16* w3h   = (_Float16*)carve(128 * 128 * sizeof(_Float16));

  k_cvt_f16<<<8, 256, 0, stream>>>(w2, w2h, 64 * 32);
  k_cvt_f16<<<64, 256, 0, stream>>>(w3, w3h, 128 * 128);
  k_mean<<<(KVOX + 255) / 256, 256, 0, stream>>>(feat, nv, mean3);

  // layer 1: 7->16 (f32 VALU) + standalone BN stats + build x32
  k_gemm1<<<(MROWS + 255) / 256, 256, 0, stream>>>(feat, mean3, w1, ybuf);
  k_stats_partial<16><<<NBSTAT, 256, 0, stream>>>(ybuf, part2);
  k_stats_final<<<1, 16, 0, stream>>>(part2, NBSTAT, 16, g1, b1, scsh1);
  k_build32<<<(KVOX + 15) / 16, 256, 0, stream>>>(ybuf, scsh1, nv, x32);

  // layer 2: 32->64 WMMA with fused stat partials
  k_gemm_wmma<32, 64><<<(int)blocks2, 256, 0, stream>>>(x32, w2h, ybuf, part1);
  k_stats_mid<<<NBSTAT, 256, 0, stream>>>(part1, blocks2, 128, part2);
  k_stats_final<<<1, 64, 0, stream>>>(part2, NBSTAT, 64, g2, b2, scsh2);
  k_build128<<<(KVOX + 7) / 8, 256, 0, stream>>>(ybuf, scsh2, nv, x128);

  // layer 3: 128->128 WMMA with fused stat partials + masked max output
  k_gemm_wmma<128, 128><<<(int)blocks3, 256, 0, stream>>>(x128, w3h, ybuf, part1);
  k_stats_mid<<<NBSTAT, 256, 0, stream>>>(part1, blocks3, 256, part2);
  k_stats_final<<<1, 128, 0, stream>>>(part2, NBSTAT, 128, g3, b3, scsh3);
  k_out<<<(KVOX + 7) / 8, 256, 0, stream>>>(ybuf, scsh3, nv, out);
}
